// InteractiveFusion_C_18064632447178
// MI455X (gfx1250) — matrix operand order repaired and checked by reading, not compiled, not run on hardware
//
#include <hip/hip_runtime.h>
#include <hip/hip_bf16.h>

typedef __bf16 bf16;
typedef __attribute__((ext_vector_type(16))) __bf16 v16bf;
typedef __attribute__((ext_vector_type(8)))  __bf16 v8bf;
typedef __attribute__((ext_vector_type(8)))  float  v8f;
typedef unsigned int u32;
typedef __attribute__((ext_vector_type(4))) unsigned int u32x4;
typedef __attribute__((ext_vector_type(8))) int i32x8;
typedef __attribute__((ext_vector_type(4))) int i32x4;

#define BATCH 8
#define CCH   256
#define HH    48
#define WW    48
#define HWPX  (HH*WW)        // 2304
#define CG    64

// ---------------------------------------------------------------------------
// WMMA fragment loaders (CDNA5 16x16x32 bf16 layouts, ISA 7.12.2)
// ---------------------------------------------------------------------------
__device__ __forceinline__ v16bf load_frag_a(const bf16* __restrict__ base, int ld,
                                             int rowBase, int kBase, int lane) {
  const int m  = lane & 15;
  const int hi = lane >> 4;
  const bf16* p = base + (long)(rowBase + m) * ld + kBase + hi * 8;
  v8bf lo = *(const v8bf*)(p);
  v8bf hv = *(const v8bf*)(p + 16);
  v16bf r;
#pragma unroll
  for (int i = 0; i < 8; ++i) { r[i] = lo[i]; r[i + 8] = hv[i]; }
  return r;
}

__device__ __forceinline__ v16bf load_frag_b(const bf16* __restrict__ bt, int ld,
                                             int nBase, int kBase, int lane) {
  const int n  = lane & 15;
  const int hi = lane >> 4;
  const bf16* p = bt + (long)(nBase + n) * ld + kBase + hi * 8;
  v8bf lo = *(const v8bf*)(p);
  v8bf hv = *(const v8bf*)(p + 16);
  v16bf r;
#pragma unroll
  for (int i = 0; i < 8; ++i) { r[i] = lo[i]; r[i + 8] = hv[i]; }
  return r;
}

__device__ __forceinline__ v8f wmma_bf16(v16bf a, v16bf b, v8f c) {
  return __builtin_amdgcn_wmma_f32_16x16x32_bf16(false, a, false, b, (short)0, c,
                                                 false, false);
}

// ---------------------------------------------------------------------------
// Tensor Data Mover: 2-D tile (bf16) global -> LDS.  D# per ISA 8.3/8.4:
//  g0: count=1 | lds_addr | global_addr[56:0] | type=2
//  g1: data_size=1(2B) | tensor_dim0/1 | tile_dim0/1 | tensor_dim0_stride
// ---------------------------------------------------------------------------
__device__ __forceinline__ void tdm_load_2d_bf16(u32 ldsAddr, const bf16* gptr,
                                                 u32 tensor_d0, u32 tensor_d1,
                                                 u32 tile_d0, u32 tile_d1,
                                                 u32 stride0_elems) {
  unsigned long long ga = (unsigned long long)(uintptr_t)gptr;
  u32x4 g0 = {};
  g0[0] = 1u;                                         // count=1 (valid user D#)
  g0[1] = ldsAddr;                                    // LDS byte address
  g0[2] = (u32)(ga & 0xffffffffu);                    // global_addr[31:0]
  g0[3] = (u32)((ga >> 32) & 0x01ffffffu) | 0x80000000u;  // [56:32] | type=2
  i32x8 g1 = {};
  g1[0] = (int)(1u << 16);                            // data_size=1 -> 2 bytes
  g1[1] = (int)((tensor_d0 & 0xffffu) << 16);         // tensor_dim0[15:0]
  g1[2] = (int)(((tensor_d0 >> 16) & 0xffffu) |       // tensor_dim0[31:16]
                ((tensor_d1 & 0xffffu) << 16));       // tensor_dim1[15:0]
  g1[3] = (int)(((tensor_d1 >> 16) & 0xffffu) |       // tensor_dim1[31:16]
                ((tile_d0 & 0xffffu) << 16));         // tile_dim0
  g1[4] = (int)(tile_d1 & 0xffffu);                   // tile_dim1 (tile_dim2=0)
  g1[5] = (int)stride0_elems;                         // tensor_dim0_stride[31:0]
  g1[6] = 0;
  g1[7] = 0;
  i32x4 z4 = {};
#if __has_include(<hip/amd_detail/amd_gfx1250_TDM.h>)
  i32x8 z8 = {};
  __builtin_amdgcn_tensor_load_to_lds(g0, g1, z4, z4, z8, 0);
#else
  __builtin_amdgcn_tensor_load_to_lds(g0, g1, z4, z4, 0);
#endif
}

// ---------------------------------------------------------------------------
// f32 -> bf16 convert (weights)
// ---------------------------------------------------------------------------
__global__ void cvt_f32_bf16(const float* __restrict__ s, bf16* __restrict__ d, int n) {
  int i = blockIdx.x * blockDim.x + threadIdx.x;
  if (i < n) d[i] = (bf16)s[i];
}

// ---------------------------------------------------------------------------
// x [B][C][HW] f32 -> xt [B][HW][C] bf16
// ---------------------------------------------------------------------------
__global__ void transpose_bf16(const float* __restrict__ x, bf16* __restrict__ xt) {
  int b = blockIdx.z;
  long idx = (long)blockIdx.x * blockDim.x + threadIdx.x;
  if (idx >= (long)CCH * HWPX) return;
  int c = (int)(idx / HWPX), n = (int)(idx % HWPX);
  xt[(long)b * HWPX * CCH + (long)n * CCH + c] = (bf16)x[(long)b * CCH * HWPX + idx];
}

// ---------------------------------------------------------------------------
// Generic WMMA GEMM, software-pipelined K loop (double-buffered fragments):
// D[m][n] = sum_k A[m][k] * BT[n][k]  (+ bias[m])
// ---------------------------------------------------------------------------
__global__ void gemm_bt_wmma(const bf16* __restrict__ A, long aStride, int lda,
                             const bf16* __restrict__ BT, long bStride, int ldb,
                             bf16* __restrict__ D, long dStride, int ldd,
                             int M, int N, int K, const float* __restrict__ bias) {
  const int lane = threadIdx.x & 31;
  const int tilesN = N >> 4;
  const int tileId = blockIdx.x * (blockDim.x >> 5) + (threadIdx.x >> 5);
  if (tileId >= (M >> 4) * tilesN) return;
  const int tm = tileId / tilesN, tn = tileId % tilesN;
  const int b = blockIdx.y;
  const bf16* Ab = A + (long)b * aStride;
  const bf16* Bb = BT + (long)b * bStride;
  bf16* Db = D + (long)b * dStride;

  v8f acc = {};
  v16bf a0 = load_frag_a(Ab, lda, tm * 16, 0, lane);
  v16bf b0 = load_frag_b(Bb, ldb, tn * 16, 0, lane);
  for (int k = 0; k < K; k += 32) {
    v16bf a1 = a0, b1 = b0;
    if (k + 32 < K) {                 // issue next-iteration loads first
      a1 = load_frag_a(Ab, lda, tm * 16, k + 32, lane);
      b1 = load_frag_b(Bb, ldb, tn * 16, k + 32, lane);
      if (k + 96 < K)
        __builtin_prefetch(Ab + (long)(tm * 16 + (lane & 15)) * lda + k + 96, 0, 1);
    }
    acc = wmma_bf16(a0, b0, acc);     // consumes oldest loads only
    a0 = a1; b0 = b1;
  }
  const int n = tn * 16 + (lane & 15);
  const int mBase = tm * 16 + (lane >> 4) * 8;
#pragma unroll
  for (int i = 0; i < 8; ++i) {
    float v = acc[i];
    if (bias) v += bias[mBase + i];
    Db[(long)(mBase + i) * ldd + n] = (bf16)v;
  }
}

// ---------------------------------------------------------------------------
// Flash attention. Block = 8 waves = 8 consecutive 16-query tiles of one
// (batch, direction). Wave 0 TDM-stages the shared 32-key K block (32x256)
// and V block (256x32) into LDS once per iteration; all waves consume via DS.
// Q fragments live in registers for the whole key loop.
// ---------------------------------------------------------------------------
__global__ void flash_attn_wmma(const bf16* __restrict__ Q1, const bf16* __restrict__ K1,
                                const bf16* __restrict__ V1,
                                const bf16* __restrict__ Q2, const bf16* __restrict__ K2,
                                const bf16* __restrict__ V2,
                                const float* __restrict__ x1, const float* __restrict__ x2,
                                const float* __restrict__ gamma1,
                                const float* __restrict__ gamma2,
                                float* __restrict__ att1, float* __restrict__ att2) {
  __shared__ bf16 Klds[32 * CCH];      // 16 KB: K rows m0..m0+31, [32][256]
  __shared__ bf16 Vlds[CCH * 32];      // 16 KB: V cols kb*32..+31, [256][32]
  __shared__ bf16 pst[8][16 * 32];     //  8 KB: per-wave P staging
  const int lane = threadIdx.x & 31;
  const int wv = threadIdx.x >> 5;

  const int QT = HWPX / 16;            // 144 query tiles
  const int GRP = QT / 8;              // 18 query groups per (b,dir)
  const int pair = blockIdx.x / GRP;   // 0..15 = (b,dir)
  const int qgrp = blockIdx.x % GRP;
  const int dir = pair & 1;
  const int b = pair >> 1;
  const int qt = qgrp * 8 + wv;

  const bf16* Q = (dir ? Q2 : Q1) + (long)b * HWPX * CCH;
  const bf16* Kp = (dir ? K2 : K1) + (long)b * HWPX * CCH;
  const bf16* Vp = (dir ? V2 : V1) + (long)b * CCH * HWPX;
  const float* xr = (dir ? x2 : x1) + (long)b * CCH * HWPX;
  const float g = dir ? gamma2[0] : gamma1[0];
  float* att = (dir ? att2 : att1) + (long)b * CCH * HWPX;

  const int qBase = qt * 16;
  const int hi = lane >> 4;
  const u32 kOff = (u32)(uintptr_t)&Klds[0];
  const u32 vOff = (u32)(uintptr_t)&Vlds[0];

  // hoist Q tile into registers: 8 fragments cover K=256
  v16bf qfrag[8];
#pragma unroll
  for (int t = 0; t < 8; ++t) qfrag[t] = load_frag_a(Q, CCH, qBase, t * 32, lane);

  v8f o[16];
#pragma unroll
  for (int t = 0; t < 16; ++t) o[t] = (v8f){};
  float rm[8], l[8];
#pragma unroll
  for (int i = 0; i < 8; ++i) { rm[i] = -1e30f; l[i] = 0.f; }

  for (int kb = 0; kb < HWPX / 32; ++kb) {
    __syncthreads();                   // previous block fully consumed
    if (wv == 0) {
      // K rows [kb*32, kb*32+32) x 256  ->  Klds[32][256]
      tdm_load_2d_bf16(kOff, Kp + (long)kb * 32 * CCH, CCH, HWPX, CCH, 32, CCH);
      // V[256][HW] columns [kb*32, kb*32+32) -> Vlds[256][32]
      tdm_load_2d_bf16(vOff, Vp + kb * 32, HWPX, CCH, 32, CCH, HWPX);
      __builtin_amdgcn_s_wait_tensorcnt(0);
    }
    __syncthreads();                   // staged data visible to all waves

    v8f s0 = {}, s1 = {};
#pragma unroll
    for (int k = 0; k < CCH; k += 32) {
      s0 = wmma_bf16(qfrag[k >> 5], load_frag_b(Klds, CCH, 0, k, lane), s0);
      s1 = wmma_bf16(qfrag[k >> 5], load_frag_b(Klds, CCH, 16, k, lane), s1);
    }
    // ---- online softmax (rows q = elem + 8*hi; key cols across lanes in half)
    float bm[8];
#pragma unroll
    for (int i = 0; i < 8; ++i) bm[i] = fmaxf(s0[i], s1[i]);
#pragma unroll
    for (int off = 8; off >= 1; off >>= 1)
#pragma unroll
      for (int i = 0; i < 8; ++i) bm[i] = fmaxf(bm[i], __shfl_xor(bm[i], off, 16));

    float sc[8], p0[8], p1[8];
#pragma unroll
    for (int i = 0; i < 8; ++i) {
      float mn = fmaxf(rm[i], bm[i]);
      sc[i] = __expf(rm[i] - mn);
      rm[i] = mn;
      p0[i] = __expf(s0[i] - mn);
      p1[i] = __expf(s1[i] - mn);
      float ls = p0[i] + p1[i];
#pragma unroll
      for (int off = 8; off >= 1; off >>= 1) ls += __shfl_xor(ls, off, 16);
      l[i] = l[i] * sc[i] + ls;
    }
#pragma unroll
    for (int t = 0; t < 16; ++t)
#pragma unroll
      for (int i = 0; i < 8; ++i) o[t][i] *= sc[i];

    // ---- restage P (S layout -> A-fragment layout) via per-wave LDS
    {
      const int kcol = lane & 15;
#pragma unroll
      for (int i = 0; i < 8; ++i) {
        int q = hi * 8 + i;
        pst[wv][q * 32 + kcol] = (bf16)p0[i];
        pst[wv][q * 32 + kcol + 16] = (bf16)p1[i];
      }
    }
    asm volatile("s_wait_dscnt 0" ::: "memory");  // in-wave cross-lane LDS RAW

    v16bf pa;
    {
      const bf16* p = &pst[wv][(lane & 15) * 32 + hi * 8];
      v8bf lo = *(const v8bf*)(p);
      v8bf hv = *(const v8bf*)(p + 16);
#pragma unroll
      for (int i = 0; i < 8; ++i) { pa[i] = lo[i]; pa[i + 8] = hv[i]; }
    }
    // ---- O += P (16x32) * V chunk, B operand from LDS V tile [256][32]
#pragma unroll
    for (int t = 0; t < 16; ++t)
      o[t] = wmma_bf16(pa, load_frag_b(Vlds, 32, t * 16, 0, lane), o[t]);
  }

  // ---- epilogue: att[b][c][n] = g * O/l + residual
  float il[8];
#pragma unroll
  for (int i = 0; i < 8; ++i) il[i] = 1.0f / l[i];
#pragma unroll
  for (int t = 0; t < 16; ++t) {
    const int c = t * 16 + (lane & 15);
#pragma unroll
    for (int i = 0; i < 8; ++i) {
      const int n = qBase + hi * 8 + i;
      const long off = (long)c * HWPX + n;
      att[off] = g * o[t][i] * il[i] + xr[off];
    }
  }
}

// ---------------------------------------------------------------------------
// conv3x3 (512 -> 64) + relu
// ---------------------------------------------------------------------------
__global__ void conv1_relu(const float* __restrict__ att1, const float* __restrict__ att2,
                           const float* __restrict__ fw1, const float* __restrict__ fb1,
                           float* __restrict__ h) {
  int idx = blockIdx.x * blockDim.x + threadIdx.x;
  const int total = BATCH * CG * HWPX;
  if (idx >= total) return;
  const int n = idx % HWPX;
  const int oc = (idx / HWPX) % CG;
  const int b = idx / (HWPX * CG);
  const int y = n / WW, x = n % WW;
  float acc = fb1[oc];
  for (int s = 0; s < 2; ++s) {
    const float* src = s ? att2 : att1;
    for (int ci = 0; ci < CCH; ++ci) {
      const float* ch = src + ((long)b * CCH + ci) * HWPX;
      const float* w = fw1 + ((long)oc * 2 * CCH + s * CCH + ci) * 9;
#pragma unroll
      for (int dy = -1; dy <= 1; ++dy) {
        int yy = y + dy;
        if (yy < 0 || yy >= HH) continue;
#pragma unroll
        for (int dx = -1; dx <= 1; ++dx) {
          int xx = x + dx;
          if (xx < 0 || xx >= WW) continue;
          acc += w[(dy + 1) * 3 + (dx + 1)] * ch[yy * WW + xx];
        }
      }
    }
  }
  h[idx] = fmaxf(acc, 0.f);
}

// ---------------------------------------------------------------------------
// conv3x3 (64 -> 2) + channel softmax + gated blend
// ---------------------------------------------------------------------------
__global__ void conv2_gate_blend(const float* __restrict__ h, const float* __restrict__ fw2,
                                 const float* __restrict__ fb2,
                                 const float* __restrict__ att1,
                                 const float* __restrict__ att2, float* __restrict__ out) {
  int idx = blockIdx.x * blockDim.x + threadIdx.x;
  if (idx >= BATCH * HWPX) return;
  const int n = idx % HWPX;
  const int b = idx / HWPX;
  const int y = n / WW, x = n % WW;
  float g0 = fb2[0], g1 = fb2[1];
  for (int oc = 0; oc < CG; ++oc) {
    const float* ch = h + ((long)b * CG + oc) * HWPX;
#pragma unroll
    for (int dy = -1; dy <= 1; ++dy) {
      int yy = y + dy;
      if (yy < 0 || yy >= HH) continue;
#pragma unroll
      for (int dx = -1; dx <= 1; ++dx) {
        int xx = x + dx;
        if (xx < 0 || xx >= WW) continue;
        float v = ch[yy * WW + xx];
        int t = (dy + 1) * 3 + (dx + 1);
        g0 += fw2[((long)0 * CG + oc) * 9 + t] * v;
        g1 += fw2[((long)1 * CG + oc) * 9 + t] * v;
      }
    }
  }
  float mx = fmaxf(g0, g1);
  float e0 = __expf(g0 - mx), e1 = __expf(g1 - mx);
  float inv = 1.0f / (e0 + e1);
  e0 *= inv;
  e1 *= inv;
  for (int c = 0; c < CCH; ++c) {
    const long off = ((long)b * CCH + c) * HWPX + n;
    out[off] = e0 * att1[off] + e1 * att2[off];
  }
}

// ---------------------------------------------------------------------------
extern "C" void kernel_launch(void* const* d_in, const int* in_sizes, int n_in,
                              void* d_out, int out_size, void* d_ws, size_t ws_size,
                              hipStream_t stream) {
  const float* x1 = (const float*)d_in[0];
  const float* x2 = (const float*)d_in[1];
  const float* wq1 = (const float*)d_in[2];
  const float* wk1 = (const float*)d_in[3];
  const float* wv1 = (const float*)d_in[4];
  const float* bv1 = (const float*)d_in[5];
  const float* gamma1 = (const float*)d_in[6];
  const float* wq2 = (const float*)d_in[7];
  const float* wk2 = (const float*)d_in[8];
  const float* wv2 = (const float*)d_in[9];
  const float* bv2 = (const float*)d_in[10];
  const float* gamma2 = (const float*)d_in[11];
  const float* fw1 = (const float*)d_in[12];
  const float* fb1 = (const float*)d_in[13];
  const float* fw2 = (const float*)d_in[14];
  const float* fb2 = (const float*)d_in[15];
  float* out = (float*)d_out;

  // ---- workspace carve
  char* p = (char*)d_ws;
  auto carve = [&](size_t bytes) -> void* {
    void* r = (void*)p;
    p += (bytes + 255) & ~(size_t)255;
    return r;
  };
  const size_t WSZ = (size_t)CCH * CCH * sizeof(bf16);
  const size_t TSZ = (size_t)BATCH * HWPX * CCH * sizeof(bf16);
  const size_t ASZ = (size_t)BATCH * CCH * HWPX * sizeof(float);
  bf16* wq1b = (bf16*)carve(WSZ);
  bf16* wk1b = (bf16*)carve(WSZ);
  bf16* wv1b = (bf16*)carve(WSZ);
  bf16* wq2b = (bf16*)carve(WSZ);
  bf16* wk2b = (bf16*)carve(WSZ);
  bf16* wv2b = (bf16*)carve(WSZ);
  bf16* Xt1 = (bf16*)carve(TSZ);
  bf16* Xt2 = (bf16*)carve(TSZ);
  bf16* Q1 = (bf16*)carve(TSZ);
  bf16* K1 = (bf16*)carve(TSZ);
  bf16* V1 = (bf16*)carve(TSZ);
  bf16* Q2 = (bf16*)carve(TSZ);
  bf16* K2 = (bf16*)carve(TSZ);
  bf16* V2 = (bf16*)carve(TSZ);
  float* att1 = (float*)carve(ASZ);
  float* att2 = (float*)carve(ASZ);
  float* hbuf = (float*)carve((size_t)BATCH * CG * HWPX * sizeof(float));

  // ---- stage 1: weight converts + input transposes
  {
    int n = CCH * CCH;
    cvt_f32_bf16<<<(n + 255) / 256, 256, 0, stream>>>(wq1, wq1b, n);
    cvt_f32_bf16<<<(n + 255) / 256, 256, 0, stream>>>(wk1, wk1b, n);
    cvt_f32_bf16<<<(n + 255) / 256, 256, 0, stream>>>(wv1, wv1b, n);
    cvt_f32_bf16<<<(n + 255) / 256, 256, 0, stream>>>(wq2, wq2b, n);
    cvt_f32_bf16<<<(n + 255) / 256, 256, 0, stream>>>(wk2, wk2b, n);
    cvt_f32_bf16<<<(n + 255) / 256, 256, 0, stream>>>(wv2, wv2b, n);
    dim3 g((CCH * HWPX + 255) / 256, 1, BATCH);
    transpose_bf16<<<g, 256, 0, stream>>>(x1, Xt1);
    transpose_bf16<<<g, 256, 0, stream>>>(x2, Xt2);
  }

  // ---- stage 2: Q/K/V projection GEMMs (WMMA, pipelined)
  const long tb = (long)HWPX * CCH;
  {
    dim3 g(288, BATCH);
    gemm_bt_wmma<<<g, 256, 0, stream>>>(Xt1, tb, CCH, wq1b, 0, CCH, Q1, tb, CCH,
                                        HWPX, CCH, CCH, nullptr);
    gemm_bt_wmma<<<g, 256, 0, stream>>>(Xt2, tb, CCH, wk1b, 0, CCH, K1, tb, CCH,
                                        HWPX, CCH, CCH, nullptr);
    gemm_bt_wmma<<<g, 256, 0, stream>>>(wv1b, 0, CCH, Xt2, tb, CCH, V1, tb, HWPX,
                                        CCH, HWPX, CCH, bv1);
    gemm_bt_wmma<<<g, 256, 0, stream>>>(Xt2, tb, CCH, wq2b, 0, CCH, Q2, tb, CCH,
                                        HWPX, CCH, CCH, nullptr);
    gemm_bt_wmma<<<g, 256, 0, stream>>>(Xt1, tb, CCH, wk2b, 0, CCH, K2, tb, CCH,
                                        HWPX, CCH, CCH, nullptr);
    gemm_bt_wmma<<<g, 256, 0, stream>>>(wv2b, 0, CCH, Xt1, tb, CCH, V2, tb, HWPX,
                                        CCH, HWPX, CCH, bv2);
  }

  // ---- stage 3: flash attention with TDM-staged K/V blocks
  {
    int blocks = BATCH * 2 * (HWPX / 16) / 8;  // 288, exact
    flash_attn_wmma<<<blocks, 256, 0, stream>>>(Q1, K1, V1, Q2, K2, V2, x1, x2,
                                                gamma1, gamma2, att1, att2);
  }

  // ---- stage 4: gate convs + blend
  {
    int n1 = BATCH * CG * HWPX;
    conv1_relu<<<(n1 + 255) / 256, 256, 0, stream>>>(att1, att2, fw1, fb1, hbuf);
    int n2 = BATCH * HWPX;
    conv2_gate_blend<<<(n2 + 255) / 256, 256, 0, stream>>>(hbuf, fw2, fb2, att1,
                                                           att2, out);
  }
}
